// PreTrainedGNN_73306501808602
// MI455X (gfx1250) — matrix operand order repaired and checked by reading, not compiled, not run on hardware
//
#include <hip/hip_runtime.h>
#include <hip/hip_bf16.h>

// ---------- types for CDNA5 WMMA ----------
typedef __attribute__((ext_vector_type(16))) __bf16         v16bf;
typedef __attribute__((ext_vector_type(16))) unsigned short v16u;
typedef __attribute__((ext_vector_type(8)))  float          v8f;

__device__ __forceinline__ unsigned short f2bf(float f) {
    // round-to-nearest-even fp32 -> bf16 (inputs are finite gaussians)
    unsigned int u = __float_as_uint(f);
    u += 0x7FFFu + ((u >> 16) & 1u);
    return (unsigned short)(u >> 16);
}

// ---------- generic zero ----------
__global__ void zero_f32(float* __restrict__ p, long long n) {
    long long t = (long long)blockIdx.x * blockDim.x + threadIdx.x;
    for (; t < n; t += (long long)gridDim.x * blockDim.x) p[t] = 0.0f;
}

// ---------- degree (in-degree at dst) ----------
__global__ void deg_kernel(const long long* __restrict__ dst, float* __restrict__ deg,
                           long long E) {
    long long e = (long long)blockIdx.x * blockDim.x + threadIdx.x;
    if (e < E) atomicAdd(&deg[(int)dst[e]], 1.0f);
}

// deg -> dinv = rsqrt(deg + 1)  (self-loop counted)
__global__ void dinv_kernel(float* __restrict__ deg, int N) {
    int i = blockIdx.x * blockDim.x + threadIdx.x;
    if (i < N) deg[i] = rsqrtf(deg[i] + 1.0f);
}

// ---------- bf16 WMMA GEMM:  C[M x 128] = A[M x K] @ W[K x 128] ----------
// one wave -> 16 rows x 128 cols (8 WMMA tiles), W chunked through LDS as bf16
template <int K>
__global__ void __launch_bounds__(256)
gemm_bf16_wmma(const float* __restrict__ A, const float* __restrict__ W,
               float* __restrict__ C, int M) {
    constexpr int KC  = 64;    // K-chunk staged in LDS
    constexpr int LDB = 132;   // padded bf16 row stride (2-dword bank rotation)
    __shared__ unsigned short sW[KC * LDB];

    const int tid  = threadIdx.x;
    const int wid  = tid >> 5;
    const int lane = tid & 31;
    const int lh   = lane >> 4;    // half-wave (selects K sub-pattern / M+8)
    const int ml   = lane & 15;

    const int tileM  = blockIdx.x * 8 + wid;
    const int mtiles = M >> 4;
    const bool active = (tileM < mtiles);

    const float* ar = A + (size_t)(active ? tileM : 0) * 16 * K + (size_t)ml * K;

    v8f acc[8] = {};

    for (int chunk = 0; chunk < K / KC; ++chunk) {
        // cooperative, coalesced stage of W chunk as bf16 into LDS
        for (int i = tid; i < KC * 128; i += 256) {
            int r = i >> 7, c = i & 127;
            sW[r * LDB + c] = f2bf(W[(size_t)(chunk * KC + r) * 128 + c]);
        }
        __syncthreads();

        if (active) {
            const int kbase = chunk * KC;
#pragma unroll
            for (int ks = 0; ks < KC / 32; ++ks) {
                const int k0 = kbase + ks * 32;
                // A fragment per ISA 16-bit A layout:
                // lanes 0-15: k0+0..7, k0+16..23 ; lanes 16-31: k0+8..15, k0+24..31
                const float* ap = ar + k0 + lh * 8;
                __builtin_prefetch(ar + k0 + KC, 0, 1);   // next chunk hint
                float4 a0 = *reinterpret_cast<const float4*>(ap);
                float4 a1 = *reinterpret_cast<const float4*>(ap + 4);
                float4 a2 = *reinterpret_cast<const float4*>(ap + 16);
                float4 a3 = *reinterpret_cast<const float4*>(ap + 20);
                v16u au;
                au[0]  = f2bf(a0.x); au[1]  = f2bf(a0.y); au[2]  = f2bf(a0.z); au[3]  = f2bf(a0.w);
                au[4]  = f2bf(a1.x); au[5]  = f2bf(a1.y); au[6]  = f2bf(a1.z); au[7]  = f2bf(a1.w);
                au[8]  = f2bf(a2.x); au[9]  = f2bf(a2.y); au[10] = f2bf(a2.z); au[11] = f2bf(a2.w);
                au[12] = f2bf(a3.x); au[13] = f2bf(a3.y); au[14] = f2bf(a3.z); au[15] = f2bf(a3.w);
                v16bf av = __builtin_bit_cast(v16bf, au);

                // B fragment: n = lane%16, k per ISA table order (matches A)
                const unsigned short* brow = &sW[(ks * 32 + lh * 8) * LDB + ml];
#pragma unroll
                for (int nt = 0; nt < 8; ++nt) {
                    v16u bu;
#pragma unroll
                    for (int e = 0; e < 16; ++e) bu[e] = brow[((e & 7) + (e >> 3) * 16) * LDB + nt * 16];
                    v16bf bv = __builtin_bit_cast(v16bf, bu);
                    acc[nt] = __builtin_amdgcn_wmma_f32_16x16x32_bf16(
                        false, av, false, bv, (short)0, acc[nt], false, false);
                }
            }
        }
        __syncthreads();
    }

    if (active) {
        // C/D layout: lanes 0-15 -> M=v, N=lane ; lanes 16-31 -> M=v+8, N=lane-16
#pragma unroll
        for (int nt = 0; nt < 8; ++nt) {
#pragma unroll
            for (int v = 0; v < 8; ++v) {
                int row = tileM * 16 + v + lh * 8;
                int col = nt * 16 + ml;
                C[(size_t)row * 128 + col] = acc[nt][v];
            }
        }
    }
}

// ---------- edge scatter: agg[dst] += h[src] * dinv[src]*dinv[dst] ----------
// one wave per edge; lane handles 4 features (float4 gather, 4 f32 atomics)
__global__ void scatter_kernel(const float* __restrict__ h,
                               const long long* __restrict__ src,
                               const long long* __restrict__ dst,
                               const float* __restrict__ dinv,
                               float* __restrict__ agg, long long E) {
    int wid  = threadIdx.x >> 5;
    int lane = threadIdx.x & 31;
    long long e = (long long)blockIdx.x * 8 + wid;
    if (e >= E) return;
    int s = (int)src[e];
    int d = (int)dst[e];
    float c = dinv[s] * dinv[d];
    float4 hv = *reinterpret_cast<const float4*>(h + (size_t)s * 128 + lane * 4);
    float* out = agg + (size_t)d * 128 + lane * 4;
    atomicAdd(out + 0, hv.x * c);
    atomicAdd(out + 1, hv.y * c);
    atomicAdd(out + 2, hv.z * c);
    atomicAdd(out + 3, hv.w * c);
}

// ---------- relu(agg + h*dinv^2 + b) -> outp ----------
__global__ void post_kernel(const float* __restrict__ agg, const float* __restrict__ h,
                            const float* __restrict__ dinv, const float* __restrict__ b,
                            float* __restrict__ outp, int N) {
    long long t = (long long)blockIdx.x * blockDim.x + threadIdx.x;
    if (t >= (long long)N * 128) return;
    int i = (int)(t >> 7);
    int f = (int)(t & 127);
    float di = dinv[i];
    float v = agg[t] + h[t] * di * di + b[f];
    outp[t] = fmaxf(v, 0.0f);
}

// ---------- per-graph max pool (relu output >= 0 -> uint atomicMax valid) ----------
__global__ void pool_kernel(const float* __restrict__ h, const long long* __restrict__ batch,
                            float* __restrict__ pool, int N) {
    long long t = (long long)blockIdx.x * blockDim.x + threadIdx.x;
    if (t >= (long long)N * 128) return;
    int i = (int)(t >> 7);
    int f = (int)(t & 127);
    int g = (int)batch[i];
    atomicMax((unsigned int*)&pool[(size_t)g * 128 + f], __float_as_uint(h[t]));
}

// ---------- head: logits = pool @ Wc + bc ; log_softmax ----------
__global__ void head_kernel(const float* __restrict__ pool, const float* __restrict__ Wc,
                            const float* __restrict__ bc, float* __restrict__ out) {
    int g = threadIdx.x;
    if (g >= 64) return;
    float l0 = bc[0], l1 = bc[1];
    const float* p = pool + (size_t)g * 128;
#pragma unroll 4
    for (int f = 0; f < 128; ++f) {
        float v = p[f];
        l0 += v * Wc[f * 2 + 0];
        l1 += v * Wc[f * 2 + 1];
    }
    float m   = fmaxf(l0, l1);
    float lse = m + logf(expf(l0 - m) + expf(l1 - m));
    out[g * 2 + 0] = l0 - lse;
    out[g * 2 + 1] = l1 - lse;
}

static inline int cdiv_ll(long long a, long long b) { return (int)((a + b - 1) / b); }

extern "C" void kernel_launch(void* const* d_in, const int* in_sizes, int n_in,
                              void* d_out, int out_size, void* d_ws, size_t ws_size,
                              hipStream_t stream) {
    const float*     x     = (const float*)d_in[0];
    const long long* edges = (const long long*)d_in[1];
    const long long* batch = (const long long*)d_in[2];
    const float*     W1    = (const float*)d_in[3];
    const float*     b1    = (const float*)d_in[4];
    const float*     W2    = (const float*)d_in[5];
    const float*     b2    = (const float*)d_in[6];
    const float*     Wc    = (const float*)d_in[7];
    const float*     bc    = (const float*)d_in[8];
    float*           out   = (float*)d_out;

    const int       N = in_sizes[0] / 512;   // 100000
    const long long E = in_sizes[1] / 2;     // 3200000
    const long long* src = edges;
    const long long* dst = edges + E;

    // workspace layout
    float*  deg    = (float*)d_ws;
    size_t  degPad = ((size_t)N + 255) & ~(size_t)255;
    float*  B0     = deg + degPad;                 // h buffer   [N*128]
    float*  B1     = B0 + (size_t)N * 128;         // agg buffer [N*128]
    float*  pool   = B1 + (size_t)N * 128;         // [64*128]
    (void)ws_size; (void)n_in; (void)out_size;

    const long long NF = (long long)N * 128;
    const int mtiles   = N >> 4;
    const int gblocks  = (mtiles + 7) / 8;

    // ---- normalization ----
    zero_f32<<<cdiv_ll(N, 256), 256, 0, stream>>>(deg, N);
    zero_f32<<<cdiv_ll(NF, 256), 256, 0, stream>>>(B1, NF);
    zero_f32<<<32, 256, 0, stream>>>(pool, 64 * 128);
    deg_kernel<<<cdiv_ll(E, 256), 256, 0, stream>>>(dst, deg, E);
    dinv_kernel<<<cdiv_ll(N, 256), 256, 0, stream>>>(deg, N);

    // ---- layer 1: h1 = x @ W1 ; agg ; relu(+self+bias) ----
    gemm_bf16_wmma<512><<<gblocks, 256, 0, stream>>>(x, W1, B0, N);
    scatter_kernel<<<cdiv_ll(E, 8), 256, 0, stream>>>(B0, src, dst, deg, B1, E);
    post_kernel<<<cdiv_ll(NF, 256), 256, 0, stream>>>(B1, B0, deg, b1, B1, N);

    // ---- layer 2: h2 = r1 @ W2 ; agg ; relu(+self+bias) ----
    gemm_bf16_wmma<128><<<gblocks, 256, 0, stream>>>(B1, W2, B0, N);
    zero_f32<<<cdiv_ll(NF, 256), 256, 0, stream>>>(B1, NF);
    scatter_kernel<<<cdiv_ll(E, 8), 256, 0, stream>>>(B0, src, dst, deg, B1, E);
    post_kernel<<<cdiv_ll(NF, 256), 256, 0, stream>>>(B1, B0, deg, b2, B0, N);

    // ---- pool + head ----
    pool_kernel<<<cdiv_ll(NF, 256), 256, 0, stream>>>(B0, batch, pool, N);
    head_kernel<<<1, 64, 0, stream>>>(pool, Wc, bc, out);
}
